// QJLSketch_9706626089276
// MI455X (gfx1250) — compile-verified, matrix operand order checked by reading, not run on hardware
//
#include <hip/hip_runtime.h>

typedef __attribute__((ext_vector_type(16))) _Float16 v16h;
typedef __attribute__((ext_vector_type(8)))  _Float16 v8h;
typedef __attribute__((ext_vector_type(8)))  float    v8f;
typedef __attribute__((ext_vector_type(4)))  float    v4f;

#define H_ 32
#define G_ 128
#define N_ 64
#define D_ 128
#define S_ 256
#define K_ 8
#define STRIDE 136   // padded f16 row stride (16B-aligned, avoids bank conflicts)

// ---- kernel 1: one-time f32 -> f16 conversion of proj_dir_quant [S,D] into workspace
__global__ void qjl_cvt_proj(const float* __restrict__ proj,
                             _Float16* __restrict__ proj16, int n) {
    int i = blockIdx.x * blockDim.x + threadIdx.x;
    if (i < n) proj16[i] = (_Float16)proj[i];
}

// ---- kernel 2: per-(h,g) sketch + hash + score
__global__ __launch_bounds__(256)
void qjl_sketch(const float* __restrict__ data,
                const float* __restrict__ query,
                const int*   __restrict__ oidx_g,
                const float* __restrict__ norm_data,
                const float* __restrict__ proj_q,     // [S,D] f32
                const _Float16* __restrict__ proj16,  // [S,D] f16 (workspace)
                float* __restrict__ out)
{
    __shared__ _Float16 data_lds[N_ * STRIDE];
    __shared__ float qf[D_];
    __shared__ float qs[S_];
    __shared__ float ip[N_];
    __shared__ int   oidx[K_];
    __shared__ float oflag[K_];

    const int tid = threadIdx.x;
    const int gid = blockIdx.x;         // flat (h,g)
    const int h   = gid >> 7;           // G_ = 128
    const size_t dataBase = (size_t)gid * (N_ * D_);

    if (tid < D_) qf[tid] = query[h * D_ + tid];
    if (tid < N_) ip[tid] = 0.f;
    if (tid < K_) oidx[tid] = oidx_g[gid * K_ + tid];

    // stage key tile f32 -> f16 into LDS (coalesced float4 reads)
    for (int i = tid; i < (N_ * D_) / 4; i += 256) {
        int row = i >> 5;
        int c4  = (i & 31) << 2;
        v4f v = *(const v4f*)(data + dataBase + row * D_ + c4);
        _Float16* p = &data_lds[row * STRIDE + c4];
        p[0] = (_Float16)v.x; p[1] = (_Float16)v.y;
        p[2] = (_Float16)v.z; p[3] = (_Float16)v.w;
    }
    __syncthreads();

    // dedup flags for outlier indices (one_hot(...).sum > 0 semantics)
    if (tid < K_) {
        float f = 1.f;
        for (int j = 0; j < tid; ++j) if (oidx[j] == oidx[tid]) f = 0.f;
        oflag[tid] = f;
    }

    // sketched query: qs[s] = sum_d q[d] * proj[s,d]   (proj_dir_score == proj_quant^T)
    {
        const float* pr = proj_q + tid * D_;
        float acc = 0.f;
        #pragma unroll 8
        for (int d = 0; d < D_; ++d) acc += qf[d] * pr[d];
        qs[tid] = acc;
    }
    __syncthreads();

    const int wave   = tid >> 5;
    const int lane   = tid & 31;
    const int laneLo = lane & 15;
    const int hi     = lane >> 4;       // K-half selector for A/B frags
    const int m0     = (wave >> 1) * 16;
    const int nh     = wave & 1;
    const int sbase  = nh * 128;

    v8f c[8], co[8];
    const v8f vzero = {};
    #pragma unroll
    for (int t = 0; t < 8; ++t) { c[t] = vzero; co[t] = vzero; }

    // ---- main GEMM: sk_full = data(64x128) x proj^T, f16 WMMA, K = 4 x 32
    const int arow = m0 + laneLo;
    #pragma unroll
    for (int k = 0; k < 4; ++k) {
        const int kb = k * 32 + hi * 8;
        union { v16h v; v8h h8[2]; } a;
        a.h8[0] = *(const v8h*)(&data_lds[arow * STRIDE + kb]);
        a.h8[1] = *(const v8h*)(&data_lds[arow * STRIDE + kb + 16]);
        #pragma unroll
        for (int t = 0; t < 8; ++t) {
            const int scol = sbase + t * 16 + laneLo;
            union { v16h v; v8h h8[2]; } b;
            b.h8[0] = *(const v8h*)(proj16 + scol * D_ + kb);
            b.h8[1] = *(const v8h*)(proj16 + scol * D_ + kb + 16);
            c[t] = __builtin_amdgcn_wmma_f32_16x16x32_f16(
                false, a.v, false, b.v, (short)0, c[t], false, false);
        }
    }

    // ---- rank-K outlier correction: co = sum_j data[:,dj] * proj[:,dj]^T
    #pragma unroll
    for (int j = 0; j < K_; ++j) {
        const int   dj = oidx[j];
        const float fj = oflag[j];
        float a[8];
        #pragma unroll
        for (int r = 0; r < 8; ++r)
            a[r] = fj * (float)data_lds[(m0 + r + hi * 8) * STRIDE + dj];
        #pragma unroll
        for (int t = 0; t < 8; ++t) {
            const int scol = sbase + t * 16 + laneLo;
            const float pd = (float)proj16[scol * D_ + dj];
            #pragma unroll
            for (int r = 0; r < 8; ++r) co[t][r] += a[r] * pd;
        }
    }

    // ---- outputs: hash packing via ballot, score via shuffle-reduce + LDS atomics
    float* out_in  = out;                                    // [gid*N+n]*32 + byte
    float* out_out = out + (size_t)H_ * G_ * N_ * 32;        // truncated outlier hash
    float* out_sc  = out + (size_t)H_ * G_ * N_ * 48;        // scores

    float ipacc[8];
    #pragma unroll
    for (int r = 0; r < 8; ++r) ipacc[r] = 0.f;

    #pragma unroll
    for (int t = 0; t < 8; ++t) {
        const int scol = sbase + t * 16 + laneLo;
        const float qv = qs[scol];
        #pragma unroll
        for (int r = 0; r < 8; ++r) {
            const float vin = c[t][r] - co[t][r];      // sk_inlier
            const bool  pin = vin > 0.f;
            ipacc[r] += pin ? qv : -qv;
            // C-tile: bits 0..15 = row m0+r (s = sbase+t*16+lane), bits 16..31 = row m0+r+8
            const unsigned mi = __builtin_amdgcn_ballot_w32(pin);
            if (lane == 0) {
                const int byteIdx = (sbase >> 3) + t * 2;
                const size_t ra = ((size_t)gid * N_ + (m0 + r)) * 32 + byteIdx;
                const size_t rb = ((size_t)gid * N_ + (m0 + r + 8)) * 32 + byteIdx;
                out_in[ra]     = (float)( mi        & 0xff);
                out_in[ra + 1] = (float)((mi >> 8)  & 0xff);
                out_in[rb]     = (float)((mi >> 16) & 0xff);
                out_in[rb + 1] = (float)((mi >> 24) & 0xff);
            }
            const unsigned mo = __builtin_amdgcn_ballot_w32(co[t][r] > 0.f);
            if (nh == 0 && lane == 0) {                // truncated: s < 128 only
                const int byteIdx = t * 2;
                const size_t ra = ((size_t)gid * N_ + (m0 + r)) * 16 + byteIdx;
                const size_t rb = ((size_t)gid * N_ + (m0 + r + 8)) * 16 + byteIdx;
                out_out[ra]     = (float)( mo        & 0xff);
                out_out[ra + 1] = (float)((mo >> 8)  & 0xff);
                out_out[rb]     = (float)((mo >> 16) & 0xff);
                out_out[rb + 1] = (float)((mo >> 24) & 0xff);
            }
        }
    }

    // reduce ip over the 16 lanes sharing a row, then combine the two s-halves
    #pragma unroll
    for (int r = 0; r < 8; ++r) {
        float v = ipacc[r];
        v += __shfl_xor(v, 1, 32);
        v += __shfl_xor(v, 2, 32);
        v += __shfl_xor(v, 4, 32);
        v += __shfl_xor(v, 8, 32);
        if (laneLo == 0) atomicAdd(&ip[m0 + r + hi * 8], v);
    }
    __syncthreads();

    if (tid < N_) {
        const float scl = 1.2533141373155003f / 256.0f;  // sqrt(pi/2)/S
        out_sc[(size_t)gid * N_ + tid] =
            scl * norm_data[(size_t)gid * N_ + tid] * ip[tid];
    }
}

extern "C" void kernel_launch(void* const* d_in, const int* in_sizes, int n_in,
                              void* d_out, int out_size, void* d_ws, size_t ws_size,
                              hipStream_t stream) {
    const float* data      = (const float*)d_in[0];
    const float* query     = (const float*)d_in[1];
    const int*   oindices  = (const int*)d_in[2];
    const float* norm_data = (const float*)d_in[3];
    // d_in[4] = norm_outlier (unused by reference outputs)
    const float* proj_q    = (const float*)d_in[5];
    // d_in[6] = proj_dir_score (== proj_q^T; we use proj_q directly)

    _Float16* proj16 = (_Float16*)d_ws;            // 64 KB

    qjl_cvt_proj<<<(S_ * D_ + 255) / 256, 256, 0, stream>>>(proj_q, proj16, S_ * D_);
    qjl_sketch<<<H_ * G_, 256, 0, stream>>>(data, query, oindices, norm_data,
                                            proj_q, proj16, (float*)d_out);
}
